// InstantNGPNetwork_80625126081123
// MI455X (gfx1250) — matrix operand order repaired and checked by reading, not compiled, not run on hardware
//
#include <hip/hip_runtime.h>
#include <hip/hip_bf16.h>

typedef __attribute__((ext_vector_type(16))) _Float16 v16h;
typedef __attribute__((ext_vector_type(8)))  float    v8f;

#define NTHREADS   128
#define NPTS_BLOCK 128
#define TBL        (1u << 19)
#define TMASK      (TBL - 1u)

__device__ __constant__ float kRes[16] = {
    16.f, 22.f, 30.f, 42.f, 58.f, 80.f, 111.f, 153.f,
    212.f, 294.f, 406.f, 561.f, 776.f, 1072.f, 1482.f, 2048.f};

// ---- WMMA helpers --------------------------------------------------------

__device__ inline v8f wmma_f16(v16h a, v16h b, v8f c) {
  // D = A(16x32,f16) * B(32x16,f16) + C(16x16,f32)
  return __builtin_amdgcn_wmma_f32_16x16x32_f16(
      /*neg_a=*/false, a, /*neg_b=*/false, b,
      /*c_mod=*/(short)0, c, /*reuse_a=*/false, /*reuse_b=*/false);
}

// Load A fragment from row-major LDS activation buffer [16 rows][pitch halfs],
// K-window of 32 starting at kbase.  Contiguous K order (matches load_Bt):
//   lanes 0-15 : V0..7 K=0..15   (for that lane's row M)
//   lanes 16-31: V0..7 K=16..31
__device__ inline v16h load_A16(const _Float16* rows, int pitch, int kbase) {
  const int lane  = threadIdx.x & 31;
  const int m     = lane & 15;
  const int khalf = lane >> 4;  // 0 or 1
  const _Float16* row = rows + m * pitch + kbase;
  v16h a;
#pragma unroll
  for (int j = 0; j < 4; ++j) {
    int k = khalf * 16 + 2 * j;         // V0..3
    a[2 * j]     = row[k];
    a[2 * j + 1] = row[k + 1];
  }
#pragma unroll
  for (int j = 0; j < 4; ++j) {
    int k = khalf * 16 + 8 + 2 * j;     // V4..7
    a[8 + 2 * j] = row[k];
    a[9 + 2 * j] = row[k + 1];
  }
  return a;
}

// Load B fragment from N-major (transposed) LDS weight buffer Wt[n][k]
// (pitchK halfs per n-row).  Contiguous K order (matches load_A16):
//   VGPR j holds K=2j,2j+1 ; lanes 0-15 -> K block 0-15, lanes 16-31 -> 16-31.
// Transposed storage makes each lane's 16 halves contiguous (-> ds_load_b128).
__device__ inline v16h load_Bt(const _Float16* wt, int pitchK, int kbase, int nbase) {
  const int lane = threadIdx.x & 31;
  const int n    = lane & 15;
  const int kh   = (lane >> 4) * 16;
  const _Float16* p = wt + (nbase + n) * pitchK + kbase + kh;
  v16h b;
#pragma unroll
  for (int j = 0; j < 16; ++j) b[j] = p[j];
  return b;
}

// Scatter a 16x16 f32 C/D tile to a row-major f16 LDS buffer with bias (+ReLU).
// C/D layout: VGPR r -> M = r + 8*(lane>=16), N = lane&15.
__device__ inline void store_act(_Float16* buf, int pitch, int rowBase, int nbase,
                                 v8f d, const float* __restrict__ bias, bool relu) {
  const int lane = threadIdx.x & 31;
  const int n    = lane & 15;
  const int ms   = (lane >> 4) * 8;
  const float bb = bias[nbase + n];
#pragma unroll
  for (int r = 0; r < 8; ++r) {
    float v = d[r] + bb;
    if (relu) v = fmaxf(v, 0.f);   // single v_max_f32 (no cmp+cndmask)
    buf[(rowBase + ms + r) * pitch + nbase + n] = (_Float16)v;
  }
}

// ---- Fused Instant-NGP forward ------------------------------------------

__global__ __launch_bounds__(NTHREADS) void ngp_fused(
    const float* __restrict__ positions, const float* __restrict__ directions,
    const float* __restrict__ table,
    const float* __restrict__ W1,  const float* __restrict__ b1,
    const float* __restrict__ W2,  const float* __restrict__ b2,
    const float* __restrict__ Wc1, const float* __restrict__ bc1,
    const float* __restrict__ Wc2, const float* __restrict__ bc2,
    const float* __restrict__ Wc3, const float* __restrict__ bc3,
    float* __restrict__ out, int B) {
  // Weights staged transposed (N-major) as f16.
  __shared__ _Float16 sWt1[64 * 32];   // W1  (32,64) -> [n][k]
  __shared__ _Float16 sWt2[16 * 64];   // W2  (64,16) -> [n][k]
  __shared__ _Float16 sWc1[64 * 32];   // Wc1 (32,64) -> [n][k]
  __shared__ _Float16 sWc2[64 * 64];   // Wc2 (64,64) -> [n][k]
  __shared__ _Float16 sWc3[16 * 64];   // Wc3 (64,3)  -> [n][k], n>=3 zero-pad
  __shared__ _Float16 bufA[NPTS_BLOCK * 64];
  __shared__ _Float16 bufB[NPTS_BLOCK * 64];

  const int tid  = threadIdx.x;
  const int lane = tid & 31;
  const int wave = tid >> 5;

  for (int i = tid; i < 32 * 64; i += NTHREADS) { int k = i >> 6, n = i & 63; sWt1[n * 32 + k] = (_Float16)W1[i]; }
  for (int i = tid; i < 64 * 16; i += NTHREADS) { int k = i >> 4, n = i & 15; sWt2[n * 64 + k] = (_Float16)W2[i]; }
  for (int i = tid; i < 32 * 64; i += NTHREADS) { int k = i >> 6, n = i & 63; sWc1[n * 32 + k] = (_Float16)Wc1[i]; }
  for (int i = tid; i < 64 * 64; i += NTHREADS) { int k = i >> 6, n = i & 63; sWc2[n * 64 + k] = (_Float16)Wc2[i]; }
  for (int i = tid; i < 16 * 64; i += NTHREADS) {
    int n = i >> 6, k = i & 63;
    sWc3[n * 64 + k] = (n < 3) ? (_Float16)Wc3[k * 3 + n] : (_Float16)0.f;
  }

  const int ptBase = blockIdx.x * NPTS_BLOCK;
  const int myPt   = ptBase + wave * 32 + lane;  // one point per lane
  const int myRow  = wave * 32 + lane;

  // ---- multi-res hash encoding: 16 levels x 8 corners, L2-resident gathers
  {
    const float px = positions[3 * myPt + 0];
    const float py = positions[3 * myPt + 1];
    const float pz = positions[3 * myPt + 2];
    const float2* tab = (const float2*)table;
#pragma unroll 1
    for (int l = 0; l < 16; ++l) {
      const float r = kRes[l];
      float sx = px * r, sy = py * r, sz = pz * r;
      float fx = floorf(sx), fy = floorf(sy), fz = floorf(sz);
      float ax = sx - fx, ay = sy - fy, az = sz - fz;
      unsigned ix = (unsigned)(int)fx, iy = (unsigned)(int)fy, iz = (unsigned)(int)fz;
      unsigned hx0 = ix,               hx1 = ix + 1u;
      unsigned hy0 = iy * 2654435761u, hy1 = hy0 + 2654435761u;
      unsigned hz0 = iz * 805459861u,  hz1 = hz0 + 805459861u;
      float e0 = 0.f, e1 = 0.f;
#pragma unroll
      for (int c = 0; c < 8; ++c) {
        const bool bx = (c >> 2) & 1, by = (c >> 1) & 1, bz = c & 1;
        unsigned h = ((bx ? hx1 : hx0) ^ (by ? hy1 : hy0) ^ (bz ? hz1 : hz0)) & TMASK;
        float2 f = tab[(size_t)l * TBL + h];
        float w = (bx ? ax : 1.f - ax) * (by ? ay : 1.f - ay) * (bz ? az : 1.f - az);
        e0 = fmaf(w, f.x, e0);
        e1 = fmaf(w, f.y, e1);
      }
      bufA[myRow * 64 + 2 * l]     = (_Float16)e0;
      bufA[myRow * 64 + 2 * l + 1] = (_Float16)e1;
    }
  }
  __syncthreads();

  // ---- density layer 1: enc(32) -> 64, ReLU   (bufA -> bufB)
#pragma unroll
  for (int t = 0; t < 2; ++t) {
    const int rowBase = wave * 32 + t * 16;
    v16h a = load_A16(&bufA[rowBase * 64], 64, 0);
#pragma unroll
    for (int nt = 0; nt < 4; ++nt) {
      v8f d = {};
      d = wmma_f16(a, load_Bt(sWt1, 32, 0, nt * 16), d);
      store_act(bufB, 64, rowBase, nt * 16, d, b1, true);
    }
  }
  __syncthreads();

  // ---- density layer 2: h(64) -> feat(16); density = exp(feat[:,0]); stage c
#pragma unroll
  for (int t = 0; t < 2; ++t) {
    const int rowBase = wave * 32 + t * 16;
    v16h a0 = load_A16(&bufB[rowBase * 64], 64, 0);
    v16h a1 = load_A16(&bufB[rowBase * 64], 64, 32);
    v8f d = {};
    d = wmma_f16(a0, load_Bt(sWt2, 64, 0, 0), d);
    d = wmma_f16(a1, load_Bt(sWt2, 64, 32, 0), d);
    const int n  = lane & 15;
    const int ms = (lane >> 4) * 8;
    if (n == 0) {  // column 0 of feat -> density
      const float bb = b2[0];
#pragma unroll
      for (int r = 0; r < 8; ++r) {
        int pt = ptBase + rowBase + ms + r;
        out[(size_t)3 * B + pt] = __expf(d[r] + bb);
      }
    }
    store_act(bufA, 64, rowBase, 0, d, b2, false);  // c[:,0:16] = feat
  }
  // SH-16 direction encoding: one point per lane -> c[:,16:32]
  {
    const float x = directions[3 * myPt + 0];
    const float y = directions[3 * myPt + 1];
    const float z = directions[3 * myPt + 2];
    const float xx = x * x, yy = y * y, zz = z * z;
    float sh[16];
    sh[0]  = 0.28209479177387814f;
    sh[1]  = 0.4886025119029199f * y;
    sh[2]  = 0.4886025119029199f * z;
    sh[3]  = 0.4886025119029199f * x;
    sh[4]  = 1.0925484305920792f * x * y;
    sh[5]  = 1.0925484305920792f * y * z;
    sh[6]  = 0.9461746957575601f * zz - 0.31539156525252f;
    sh[7]  = 1.0925484305920792f * x * z;
    sh[8]  = 0.5462742152960396f * (xx - yy);
    sh[9]  = 0.5900435899266435f * y * (3.f * xx - yy);
    sh[10] = 2.890611442640554f * x * y * z;
    sh[11] = 0.4570457994644658f * y * (5.f * zz - 1.f);
    sh[12] = 0.3731763325901154f * z * (5.f * zz - 3.f);
    sh[13] = 0.4570457994644658f * x * (5.f * zz - 1.f);
    sh[14] = 1.445305721320277f * z * (xx - yy);
    sh[15] = 0.5900435899266435f * x * (xx - 3.f * yy);
#pragma unroll
    for (int i = 0; i < 16; ++i) bufA[myRow * 64 + 16 + i] = (_Float16)sh[i];
  }
  __syncthreads();

  // ---- color layer 1: c(32) -> 64, ReLU   (bufA -> bufB)
#pragma unroll
  for (int t = 0; t < 2; ++t) {
    const int rowBase = wave * 32 + t * 16;
    v16h a = load_A16(&bufA[rowBase * 64], 64, 0);
#pragma unroll
    for (int nt = 0; nt < 4; ++nt) {
      v8f d = {};
      d = wmma_f16(a, load_Bt(sWc1, 32, 0, nt * 16), d);
      store_act(bufB, 64, rowBase, nt * 16, d, bc1, true);
    }
  }
  __syncthreads();

  // ---- color layer 2: 64 -> 64, ReLU   (bufB -> bufA)
#pragma unroll
  for (int t = 0; t < 2; ++t) {
    const int rowBase = wave * 32 + t * 16;
    v16h a0 = load_A16(&bufB[rowBase * 64], 64, 0);
    v16h a1 = load_A16(&bufB[rowBase * 64], 64, 32);
#pragma unroll
    for (int nt = 0; nt < 4; ++nt) {
      v8f d = {};
      d = wmma_f16(a0, load_Bt(sWc2, 64, 0, nt * 16), d);
      d = wmma_f16(a1, load_Bt(sWc2, 64, 32, nt * 16), d);
      store_act(bufA, 64, rowBase, nt * 16, d, bc2, true);
    }
  }
  __syncthreads();

  // ---- color layer 3: 64 -> 3 (padded to 16), sigmoid, write rgb
#pragma unroll
  for (int t = 0; t < 2; ++t) {
    const int rowBase = wave * 32 + t * 16;
    v16h a0 = load_A16(&bufA[rowBase * 64], 64, 0);
    v16h a1 = load_A16(&bufA[rowBase * 64], 64, 32);
    v8f d = {};
    d = wmma_f16(a0, load_Bt(sWc3, 64, 0, 0), d);
    d = wmma_f16(a1, load_Bt(sWc3, 64, 32, 0), d);
    const int n  = lane & 15;
    const int ms = (lane >> 4) * 8;
    if (n < 3) {
      const float bb = bc3[n];
#pragma unroll
      for (int r = 0; r < 8; ++r) {
        int pt = ptBase + rowBase + ms + r;
        float v = d[r] + bb;
        out[(size_t)pt * 3 + n] = 1.f / (1.f + __expf(-v));
      }
    }
  }
}

extern "C" void kernel_launch(void* const* d_in, const int* in_sizes, int n_in,
                              void* d_out, int out_size, void* d_ws, size_t ws_size,
                              hipStream_t stream) {
  const float* positions  = (const float*)d_in[0];
  const float* directions = (const float*)d_in[1];
  const float* table      = (const float*)d_in[2];
  const float* W1  = (const float*)d_in[3];
  const float* b1  = (const float*)d_in[4];
  const float* W2  = (const float*)d_in[5];
  const float* b2  = (const float*)d_in[6];
  const float* Wc1 = (const float*)d_in[7];
  const float* bc1 = (const float*)d_in[8];
  const float* Wc2 = (const float*)d_in[9];
  const float* bc2 = (const float*)d_in[10];
  const float* Wc3 = (const float*)d_in[11];
  const float* bc3 = (const float*)d_in[12];
  float* out = (float*)d_out;

  const int B = in_sizes[0] / 3;
  dim3 grid((B + NPTS_BLOCK - 1) / NPTS_BLOCK);
  dim3 block(NTHREADS);
  hipLaunchKernelGGL(ngp_fused, grid, block, 0, stream,
                     positions, directions, table,
                     W1, b1, W2, b2, Wc1, bc1, Wc2, bc2, Wc3, bc3, out, B);
}
